// Block_6339371729729
// MI455X (gfx1250) — matrix operand order repaired and checked by reading, not compiled, not run on hardware
//
#include <hip/hip_runtime.h>

typedef __attribute__((ext_vector_type(16))) __bf16 v16bf;
typedef __attribute__((ext_vector_type(8)))  float  v8f;
typedef __attribute__((ext_vector_type(4)))  unsigned int u32x4;
typedef __attribute__((ext_vector_type(8)))  int  i32x8;
typedef __attribute__((ext_vector_type(4)))  int  i32x4;

#define B_  4
#define T_  1024
#define C_  1024
#define H_  16
#define D_  64
#define FF_ 4096
#define E_  4
#define BT_ (B_*T_)

union Frag16 { v16bf v; uint4 q[2]; };
union Acc    { v8f v; float f[8]; };

__device__ __forceinline__ __bf16 bfc(float x) { return (__bf16)x; }

// ---------------------------------------------------------------------------
// TDM: async DMA of a 128-row x 32-elem bf16 tile (row stride K elems) from
// global into LDS with 4-DWORD padding after every 16 DWORDs -> LDS row
// stride 40 elems, matching the WMMA fragment reader layout.
// D# fields per CDNA5 ISA ch.8. 6-arg builtin form (clang-23 / therock-10.0).
// ---------------------------------------------------------------------------
__device__ __forceinline__ void tdm_tile_128x32(unsigned lds_off, const __bf16* g,
                                                int tens_rowlen, int stride_elems) {
    unsigned long long ga = (unsigned long long)(const void*)g;
    u32x4 g0;
    g0[0] = 1u;                                         // count=1, user mode
    g0[1] = lds_off;                                    // LDS byte address
    g0[2] = (unsigned)(ga & 0xFFFFFFFFu);               // global addr lo
    g0[3] = (unsigned)((ga >> 32) & 0x1FFFFFFu)         // global addr hi (57b)
          | (2u << 30);                                 // type = image
    i32x8 g1;
    g1[0] = (1 << 16)                                   // data_size = 2 bytes
          | (1 << 20)                                   // pad_enable
          | (3 << 22)                                   // pad_interval: 16 DWORDs
          | (3 << 25);                                  // pad_amount:   4 DWORDs
    g1[1] = (int)(((unsigned)tens_rowlen & 0xFFFFu) << 16);        // tensor_dim0 lo16
    g1[2] = (int)(((unsigned)tens_rowlen >> 16) & 0xFFFFu);        // t0 hi16 | t1 lo16=0
    g1[3] = (int)(0x10u | (32u << 16));                 // t1 hi16 (t1=1M) | tile_dim0=32
    g1[4] = 128;                                        // tile_dim1=128, tile_dim2=0
    g1[5] = stride_elems;                               // tensor_dim0_stride lo32
    g1[6] = 0;
    g1[7] = 0;
    i32x4 z4 = {0, 0, 0, 0};
    i32x8 z8 = {0, 0, 0, 0, 0, 0, 0, 0};
    __builtin_amdgcn_tensor_load_to_lds(g0, g1, z4, z4, z8, 0);
}

// ---------------------------------------------------------------------------
// fp32 [K,N] -> bf16 [N,K]  (transpose + downconvert), 32x32 tiles
// ---------------------------------------------------------------------------
__global__ void k_cvt_t(const float* __restrict__ src, __bf16* __restrict__ dst,
                        int K, int N) {
    __shared__ float ts[32][33];
    int n0 = blockIdx.x * 32, k0 = blockIdx.y * 32;
    int tx = threadIdx.x & 31, ty = threadIdx.x >> 5;   // 32 x 8
#pragma unroll
    for (int i = 0; i < 4; i++) {
        int r = ty + i * 8;
        ts[r][tx] = src[(long)(k0 + r) * N + n0 + tx];
    }
    __syncthreads();
#pragma unroll
    for (int i = 0; i < 4; i++) {
        int r = ty + i * 8;                              // local n
        dst[(long)(n0 + r) * K + k0 + tx] = bfc(ts[tx][r]);
    }
}

// ---------------------------------------------------------------------------
// LayerNorm: fp32 row -> bf16 row. One block (256 thr) per row of C_=1024.
// ---------------------------------------------------------------------------
__global__ void k_ln(const float* __restrict__ x, const float* __restrict__ g,
                     const float* __restrict__ bb, __bf16* __restrict__ h) {
    __shared__ float sb[256];
    int row = blockIdx.x;
    const float* xr = x + (long)row * C_;
    float s = 0.f, s2 = 0.f;
    for (int c = threadIdx.x; c < C_; c += 256) { float v = xr[c]; s += v; s2 += v * v; }
    sb[threadIdx.x] = s; __syncthreads();
    for (int st = 128; st > 0; st >>= 1) { if (threadIdx.x < st) sb[threadIdx.x] += sb[threadIdx.x + st]; __syncthreads(); }
    float mean = sb[0] / C_; __syncthreads();
    sb[threadIdx.x] = s2; __syncthreads();
    for (int st = 128; st > 0; st >>= 1) { if (threadIdx.x < st) sb[threadIdx.x] += sb[threadIdx.x + st]; __syncthreads(); }
    float var = sb[0] / C_ - mean * mean;
    float rstd = rsqrtf(var + 1e-5f);
    for (int c = threadIdx.x; c < C_; c += 256)
        h[(long)row * C_ + c] = bfc((xr[c] - mean) * rstd * g[c] + bb[c]);
}

// ---------------------------------------------------------------------------
// bf16 WMMA GEMM: out = epilogue(A[M,K] @ Bt[N,K]^T).
// Block tile 128x128, 8 waves (4Mx2N), wave tile 32x64 (2x4 of 16x16), K=32.
// Double-buffered LDS; dense tiles streamed by the Tensor Data Mover
// (tensor_load_to_lds, TENSORcnt), gathered A tiles loaded manually.
// MODE: 0 bf16   1 bf16 QK-layout [B,H,T,D]   2 bf16 V^T-layout [B,H,D,T]
//       3 f32 out = resid + bias + acc        4 bf16 out = relu(acc + bias)
//       5 bf16 QK-layout scaled by 1/sqrt(D)  (Q projection)
// ROUTE: 0 none  1 gather A rows via perm, compact out rows (expert up-proj)
//        2 compact A rows, scatter out/resid rows via perm (expert down-proj)
// ---------------------------------------------------------------------------
template <int MODE, int ROUTE>
__global__ void k_gemm(const __bf16* __restrict__ A, const __bf16* __restrict__ Bt,
                       const float* __restrict__ bias, const float* __restrict__ resid,
                       void* __restrict__ outp, int M, int N, int K,
                       const int* __restrict__ perm, const int* __restrict__ offs,
                       int expert) {
    __shared__ __bf16 As[2][128 * 40];
    __shared__ __bf16 Bs[2][128 * 40];

    int base = 0, cnt = M;
    if (ROUTE) {
        base = offs[expert];
        cnt  = offs[expert + 1] - base;
        if ((int)(blockIdx.y * 128) >= cnt) return;
    }
    int m0 = blockIdx.y * 128, n0 = blockIdx.x * 128;
    int t = threadIdx.x;
    int lane = t & 31, wid = t >> 5;
    int wm = wid >> 1, wn = wid & 1;
    int lr = lane & 15, hi = lane >> 4;

    Acc acc[2][4];
#pragma unroll
    for (int i = 0; i < 2; i++)
#pragma unroll
        for (int j = 0; j < 4; j++)
#pragma unroll
            for (int e = 0; e < 8; e++) acc[i][j].f[e] = 0.f;

    // per-thread source row pointer for gathered (manual) A tiles
    const __bf16* arp = nullptr;
    int akp = (t & 1) * 16;
    if (ROUTE) {
        int rl = m0 + (t >> 1); if (rl > cnt - 1) rl = cnt - 1;
        int ar = (ROUTE == 1) ? perm[base + rl] : (base + rl);
        arp = A + (long)ar * K + akp;
    }
    int nk = K >> 5;

#define ISSUE_TILE(IT)                                                          \
    do {                                                                        \
        int _buf = (IT) & 1;                                                    \
        if (ROUTE == 0) {                                                       \
            if (wid == 0)                                                       \
                tdm_tile_128x32((unsigned)(unsigned long long)(void*)&As[_buf][0], \
                                A + (long)m0 * K + (IT) * 32, K, K);            \
        } else {                                                                \
            const uint4* _ag = (const uint4*)(arp + (IT) * 32);                 \
            *(uint4*)&As[_buf][(t >> 1) * 40 + akp]     = _ag[0];               \
            *(uint4*)&As[_buf][(t >> 1) * 40 + akp + 8] = _ag[1];               \
        }                                                                       \
        if (wid == 0)                                                           \
            tdm_tile_128x32((unsigned)(unsigned long long)(void*)&Bs[_buf][0],  \
                            Bt + (long)n0 * K + (IT) * 32, K, K);               \
    } while (0)

    ISSUE_TILE(0);
    for (int it = 0; it < nk; ++it) {
        if (it + 1 < nk) ISSUE_TILE(it + 1);
        if (wid == 0) {
            if (it + 1 < nk) __builtin_amdgcn_s_wait_tensorcnt(ROUTE ? 1 : 2);
            else             __builtin_amdgcn_s_wait_tensorcnt(0);
        }
        __syncthreads();
        int buf = it & 1;
        Frag16 af[2], bfg[4];
#pragma unroll
        for (int i = 0; i < 2; i++) {
            const __bf16* p = &As[buf][(wm * 32 + i * 16 + lr) * 40 + hi * 8];
            af[i].q[0] = *(const uint4*)p;
            af[i].q[1] = *(const uint4*)(p + 16);
        }
#pragma unroll
        for (int j = 0; j < 4; j++) {
            const __bf16* p = &Bs[buf][(wn * 64 + j * 16 + lr) * 40 + hi * 8];
            bfg[j].q[0] = *(const uint4*)p;
            bfg[j].q[1] = *(const uint4*)(p + 16);
        }
#pragma unroll
        for (int i = 0; i < 2; i++)
#pragma unroll
            for (int j = 0; j < 4; j++)
                acc[i][j].v = __builtin_amdgcn_wmma_f32_16x16x32_bf16(
                    false, af[i].v, false, bfg[j].v, (short)0, acc[i][j].v, false, false);
        __syncthreads();
    }
#undef ISSUE_TILE

    // epilogue
#pragma unroll
    for (int i = 0; i < 2; i++)
#pragma unroll
        for (int j = 0; j < 4; j++) {
            int mbase = m0 + wm * 32 + i * 16 + hi * 8;
            int nn = n0 + wn * 64 + j * 16 + lr;
#pragma unroll
            for (int e = 0; e < 8; e++) {
                int ml = mbase + e;                 // local (compact) row when routed
                if (ROUTE && ml >= cnt) continue;
                int orow;
                if (ROUTE == 1) orow = base + ml;
                else if (ROUTE == 2) orow = perm[base + ml];
                else orow = ml;
                float v = acc[i][j].f[e];
                if (MODE == 0) {
                    ((__bf16*)outp)[(long)orow * N + nn] = bfc(v);
                } else if (MODE == 1 || MODE == 5) {
                    int b = ml >> 10, tt = ml & 1023, h = nn >> 6, d = nn & 63;
                    float sv = (MODE == 5) ? v * 0.125f : v;   // fold 1/sqrt(64) into Q
                    ((__bf16*)outp)[(((long)(b * H_ + h)) * T_ + tt) * D_ + d] = bfc(sv);
                } else if (MODE == 2) {
                    int b = ml >> 10, tt = ml & 1023, h = nn >> 6, d = nn & 63;
                    ((__bf16*)outp)[(((long)(b * H_ + h)) * D_ + d) * T_ + tt] = bfc(v);
                } else if (MODE == 3) {
                    ((float*)outp)[(long)orow * N + nn] =
                        resid[(long)orow * N + nn] + bias[nn] + v;
                } else if (MODE == 4) {
                    ((__bf16*)outp)[(long)orow * N + nn] = bfc(fmaxf(v + bias[nn], 0.f));
                }
            }
        }
}

// ---------------------------------------------------------------------------
// Flash attention (causal). q: [B,H,T,D] bf16 PRE-SCALED by 1/sqrt(D);
// k: [B,H,T,D] bf16; vt: [B,H,D,T] bf16.
// Block = 4 waves x 16 q-rows = 64 rows. Online softmax per 32-key chunk.
// Full chunks run mask-free; exactly one diagonal chunk applies the causal
// mask. Row sums come from an extra WMMA against an all-ones B fragment
// (each output element = its row's sum), eliminating the sum shuffles.
// ---------------------------------------------------------------------------
__global__ void k_attn(const __bf16* __restrict__ q, const __bf16* __restrict__ k,
                       const __bf16* __restrict__ vt, __bf16* __restrict__ ao) {
    __shared__ __bf16 pbuf[4][16 * 32];
    int w = threadIdx.x >> 5, lane = threadIdx.x & 31;
    int lr = lane & 15, hi = lane >> 4;
    int bh = blockIdx.y, b = bh >> 4, h = bh & 15;
    int q0 = blockIdx.x * 64 + w * 16;
    const __bf16* qb = q  + (long)bh * T_ * D_;
    const __bf16* kb = k  + (long)bh * T_ * D_;
    const __bf16* vb = vt + (long)bh * D_ * T_;

    Frag16 qa[2];
#pragma unroll
    for (int kk = 0; kk < 2; kk++) {
        const __bf16* p = qb + (long)(q0 + lr) * D_ + kk * 32 + hi * 8;
        qa[kk].q[0] = *(const uint4*)p;
        qa[kk].q[1] = *(const uint4*)(p + 16);
    }
    Frag16 ones;
#pragma unroll
    for (int i = 0; i < 2; i++) {
        ones.q[i].x = 0x3F803F80u; ones.q[i].y = 0x3F803F80u;   // bf16 1.0 pairs
        ones.q[i].z = 0x3F803F80u; ones.q[i].w = 0x3F803F80u;
    }

    Acc o[4];
#pragma unroll
    for (int j = 0; j < 4; j++)
#pragma unroll
        for (int e = 0; e < 8; e++) o[j].f[e] = 0.f;
    float mi[8], li[8];
#pragma unroll
    for (int e = 0; e < 8; e++) { mi[e] = -1e30f; li[e] = 0.f; }

    auto chunk = [&](int s0, bool MASK) {
        Acc sc[2];
#pragma unroll
        for (int sub = 0; sub < 2; sub++) {
#pragma unroll
            for (int e = 0; e < 8; e++) sc[sub].f[e] = 0.f;
#pragma unroll
            for (int kk = 0; kk < 2; kk++) {
                Frag16 kf;
                const __bf16* p = kb + (long)(s0 + sub * 16 + lr) * D_ + kk * 32 + hi * 8;
                kf.q[0] = *(const uint4*)p;
                kf.q[1] = *(const uint4*)(p + 16);
                sc[sub].v = __builtin_amdgcn_wmma_f32_16x16x32_bf16(
                    false, qa[kk].v, false, kf.v, (short)0, sc[sub].v, false, false);
            }
        }
        float rmax[8];
#pragma unroll
        for (int e = 0; e < 8; e++) {
            float a0 = sc[0].f[e], a1 = sc[1].f[e];
            if (MASK) {
                int tt = q0 + hi * 8 + e;
                if (s0 + lr > tt)      a0 = -1e30f;
                if (s0 + 16 + lr > tt) a1 = -1e30f;
                sc[0].f[e] = a0; sc[1].f[e] = a1;
            }
            rmax[e] = fmaxf(a0, a1);
        }
#pragma unroll
        for (int e = 0; e < 8; e++)
#pragma unroll
            for (int off = 1; off < 16; off <<= 1)
                rmax[e] = fmaxf(rmax[e], __shfl_xor(rmax[e], off, 16));
        float alpha[8];
#pragma unroll
        for (int e = 0; e < 8; e++) {
            float mn = fmaxf(mi[e], rmax[e]);
            alpha[e] = __expf(mi[e] - mn);
            mi[e] = mn;
            float p0 = __expf(sc[0].f[e] - mn);
            float p1 = __expf(sc[1].f[e] - mn);
            pbuf[w][(hi * 8 + e) * 32 + lr]      = bfc(p0);
            pbuf[w][(hi * 8 + e) * 32 + 16 + lr] = bfc(p1);
        }
        __builtin_amdgcn_wave_barrier();
        asm volatile("s_wait_dscnt 0" ::: "memory");    // P stores visible for re-stripe read

        Frag16 pf;
        {
            const __bf16* p = &pbuf[w][lr * 32 + hi * 8];
            pf.q[0] = *(const uint4*)p;
            pf.q[1] = *(const uint4*)(p + 16);
        }
        Acc sums;
        {
            v8f zc = {};
            sums.v = __builtin_amdgcn_wmma_f32_16x16x32_bf16(
                false, pf.v, false, ones.v, (short)0, zc, false, false);
        }
#pragma unroll
        for (int jd = 0; jd < 4; jd++) {
#pragma unroll
            for (int e = 0; e < 8; e++) o[jd].f[e] *= alpha[e];
            Frag16 vf;
            const __bf16* p = vb + (long)(jd * 16 + lr) * T_ + s0 + hi * 8;
            vf.q[0] = *(const uint4*)p;
            vf.q[1] = *(const uint4*)(p + 16);
            o[jd].v = __builtin_amdgcn_wmma_f32_16x16x32_bf16(
                false, pf.v, false, vf.v, (short)0, o[jd].v, false, false);
        }
#pragma unroll
        for (int e = 0; e < 8; e++)
            li[e] = li[e] * alpha[e] + sums.f[e];
    };

    int s0 = 0;
    for (; s0 + 31 <= q0; s0 += 32) chunk(s0, false);   // fully unmasked chunks
    for (; s0 < q0 + 16; s0 += 32) chunk(s0, true);     // single diagonal chunk

#pragma unroll
    for (int jd = 0; jd < 4; jd++)
#pragma unroll
        for (int e = 0; e < 8; e++) {
            int tt = q0 + hi * 8 + e;
            int c = h * 64 + jd * 16 + lr;
            ao[((long)(b * T_) + tt) * C_ + c] = bfc(o[jd].f[e] / li[e]);
        }
}

// ---------------------------------------------------------------------------
// Gate: per-token logits over E_=4 experts, argmax (K=1 => weight 1.0).
// One wave per token.
// ---------------------------------------------------------------------------
__global__ void k_gate(const __bf16* __restrict__ h3, const float* __restrict__ gw,
                       const float* __restrict__ gb, int* __restrict__ eid,
                       int* __restrict__ cnt) {
    int w = threadIdx.x >> 5, lane = threadIdx.x & 31;
    int tok = blockIdx.x * 8 + w;
    const __bf16* hr = h3 + (long)tok * C_;
    float a[E_] = {0.f, 0.f, 0.f, 0.f};
    for (int c = lane; c < C_; c += 32) {
        float hv = (float)hr[c];
#pragma unroll
        for (int e = 0; e < E_; e++) a[e] += hv * gw[c * E_ + e];
    }
#pragma unroll
    for (int e = 0; e < E_; e++)
#pragma unroll
        for (int off = 16; off > 0; off >>= 1) a[e] += __shfl_xor(a[e], off, 32);
    if (lane == 0) {
        int best = 0; float bv = a[0] + gb[0];
        for (int e = 1; e < E_; e++) { float v = a[e] + gb[e]; if (v > bv) { bv = v; best = e; } }
        eid[tok] = best;
        atomicAdd(&cnt[best], 1);
    }
}

__global__ void k_init(int* cnt, int* cur) {
    if (threadIdx.x < E_) { cnt[threadIdx.x] = 0; cur[threadIdx.x] = 0; }
}
__global__ void k_scan(const int* cnt, int* offs) {
    offs[0] = 0;
    for (int e = 0; e < E_; e++) offs[e + 1] = offs[e] + cnt[e];
}
__global__ void k_scatter(const int* eid, const int* offs, int* cur, int* perm) {
    int tok = blockIdx.x * 256 + threadIdx.x;
    int e = eid[tok];
    int slot = offs[e] + atomicAdd(&cur[e], 1);
    perm[slot] = tok;
}

// ---------------------------------------------------------------------------
extern "C" void kernel_launch(void* const* d_in, const int* in_sizes, int n_in,
                              void* d_out, int out_size, void* d_ws, size_t ws_size,
                              hipStream_t stream) {
    const float* x      = (const float*)d_in[0];
    const float* wq     = (const float*)d_in[1];
    const float* wk     = (const float*)d_in[2];
    const float* wv     = (const float*)d_in[3];
    const float* wo     = (const float*)d_in[4];
    const float* bo     = (const float*)d_in[5];
    const float* ff_w1  = (const float*)d_in[6];
    const float* ff_b1  = (const float*)d_in[7];
    const float* ff_w2  = (const float*)d_in[8];
    const float* ff_b2  = (const float*)d_in[9];
    const float* exp_w1 = (const float*)d_in[10];
    const float* exp_b1 = (const float*)d_in[11];
    const float* exp_w2 = (const float*)d_in[12];
    const float* exp_b2 = (const float*)d_in[13];
    const float* gate_w = (const float*)d_in[14];
    const float* gate_b = (const float*)d_in[15];
    const float* ln1_g  = (const float*)d_in[16];
    const float* ln1_b  = (const float*)d_in[17];
    const float* ln2_g  = (const float*)d_in[18];
    const float* ln2_b  = (const float*)d_in[19];
    const float* ln3_g  = (const float*)d_in[20];
    const float* ln3_b  = (const float*)d_in[21];
    float* out = (float*)d_out;

    char* ws = (char*)d_ws;
    size_t off = 0;
    auto alloc = [&](size_t bytes) -> void* {
        void* p = ws + off; off += (bytes + 255) & ~(size_t)255; return p;
    };
    __bf16* wqT   = (__bf16*)alloc((size_t)C_ * C_ * 2);
    __bf16* wkT   = (__bf16*)alloc((size_t)C_ * C_ * 2);
    __bf16* wvT   = (__bf16*)alloc((size_t)C_ * C_ * 2);
    __bf16* woT   = (__bf16*)alloc((size_t)C_ * C_ * 2);
    __bf16* fw1T  = (__bf16*)alloc((size_t)C_ * FF_ * 2);
    __bf16* fw2T  = (__bf16*)alloc((size_t)FF_ * C_ * 2);
    __bf16* ew1T  = (__bf16*)alloc((size_t)E_ * C_ * FF_ * 2);
    __bf16* ew2T  = (__bf16*)alloc((size_t)E_ * FF_ * C_ * 2);
    __bf16* hbuf  = (__bf16*)alloc((size_t)BT_ * C_ * 2);   // ln outputs (reused)
    __bf16* qb    = (__bf16*)alloc((size_t)BT_ * C_ * 2);
    __bf16* kb    = (__bf16*)alloc((size_t)BT_ * C_ * 2);
    __bf16* vtb   = (__bf16*)alloc((size_t)BT_ * C_ * 2);
    __bf16* aob   = (__bf16*)alloc((size_t)BT_ * C_ * 2);
    float*  x1    = (float*)alloc((size_t)BT_ * C_ * 4);
    float*  x2    = (float*)alloc((size_t)BT_ * C_ * 4);
    __bf16* hf    = (__bf16*)alloc((size_t)BT_ * FF_ * 2);  // ffn hidden / expert hidden
    int* cnt  = (int*)alloc(E_ * 4);
    int* cur  = (int*)alloc(E_ * 4);
    int* offs = (int*)alloc((E_ + 1) * 4);
    int* eid  = (int*)alloc(BT_ * 4);
    int* perm = (int*)alloc(BT_ * 4);

    // ---- weight convert + transpose to bf16 ----
    k_cvt_t<<<dim3(C_ / 32, C_ / 32), 256, 0, stream>>>(wq, wqT, C_, C_);
    k_cvt_t<<<dim3(C_ / 32, C_ / 32), 256, 0, stream>>>(wk, wkT, C_, C_);
    k_cvt_t<<<dim3(C_ / 32, C_ / 32), 256, 0, stream>>>(wv, wvT, C_, C_);
    k_cvt_t<<<dim3(C_ / 32, C_ / 32), 256, 0, stream>>>(wo, woT, C_, C_);
    k_cvt_t<<<dim3(FF_ / 32, C_ / 32), 256, 0, stream>>>(ff_w1, fw1T, C_, FF_);
    k_cvt_t<<<dim3(C_ / 32, FF_ / 32), 256, 0, stream>>>(ff_w2, fw2T, FF_, C_);
    for (int e = 0; e < E_; e++) {
        k_cvt_t<<<dim3(FF_ / 32, C_ / 32), 256, 0, stream>>>(
            exp_w1 + (size_t)e * C_ * FF_, ew1T + (size_t)e * FF_ * C_, C_, FF_);
        k_cvt_t<<<dim3(C_ / 32, FF_ / 32), 256, 0, stream>>>(
            exp_w2 + (size_t)e * FF_ * C_, ew2T + (size_t)e * C_ * FF_, FF_, C_);
    }

    // ---- attention ----
    k_ln<<<BT_, 256, 0, stream>>>(x, ln1_g, ln1_b, hbuf);
    k_gemm<5, 0><<<dim3(C_ / 128, BT_ / 128), 256, 0, stream>>>(
        hbuf, wqT, nullptr, nullptr, qb, BT_, C_, C_, nullptr, nullptr, 0);
    k_gemm<1, 0><<<dim3(C_ / 128, BT_ / 128), 256, 0, stream>>>(
        hbuf, wkT, nullptr, nullptr, kb, BT_, C_, C_, nullptr, nullptr, 0);
    k_gemm<2, 0><<<dim3(C_ / 128, BT_ / 128), 256, 0, stream>>>(
        hbuf, wvT, nullptr, nullptr, vtb, BT_, C_, C_, nullptr, nullptr, 0);
    k_attn<<<dim3(T_ / 64, B_ * H_), 128, 0, stream>>>(qb, kb, vtb, aob);
    k_gemm<3, 0><<<dim3(C_ / 128, BT_ / 128), 256, 0, stream>>>(
        aob, woT, bo, x, x1, BT_, C_, C_, nullptr, nullptr, 0);

    // ---- dense FFN ----
    k_ln<<<BT_, 256, 0, stream>>>(x1, ln2_g, ln2_b, hbuf);
    k_gemm<4, 0><<<dim3(FF_ / 128, BT_ / 128), 256, 0, stream>>>(
        hbuf, fw1T, ff_b1, nullptr, hf, BT_, FF_, C_, nullptr, nullptr, 0);
    k_gemm<3, 0><<<dim3(C_ / 128, BT_ / 128), 256, 0, stream>>>(
        hf, fw2T, ff_b2, x1, x2, BT_, C_, FF_, nullptr, nullptr, 0);

    // ---- MoE: gate -> route -> grouped expert GEMMs (top-1 => weight 1.0) ----
    k_ln<<<BT_, 256, 0, stream>>>(x2, ln3_g, ln3_b, hbuf);
    k_init<<<1, 32, 0, stream>>>(cnt, cur);
    k_gate<<<BT_ / 8, 256, 0, stream>>>(hbuf, gate_w, gate_b, eid, cnt);
    k_scan<<<1, 1, 0, stream>>>(cnt, offs);
    k_scatter<<<BT_ / 256, 256, 0, stream>>>(eid, offs, cur, perm);
    for (int e = 0; e < E_; e++) {
        k_gemm<4, 1><<<dim3(FF_ / 128, BT_ / 128), 256, 0, stream>>>(
            hbuf, ew1T + (size_t)e * FF_ * C_, exp_b1 + (size_t)e * FF_,
            nullptr, hf, BT_, FF_, C_, perm, offs, e);
        k_gemm<3, 2><<<dim3(C_ / 128, BT_ / 128), 256, 0, stream>>>(
            hf, ew2T + (size_t)e * C_ * FF_, exp_b2 + (size_t)e * C_,
            x2, out, BT_, C_, FF_, perm, offs, e);
    }
}
